// AttentionHead_59176059404569
// MI455X (gfx1250) — compile-verified, matrix operand order checked by reading, not run on hardware
//
#include <hip/hip_runtime.h>

#define BB 4
#define SS 4096
#define DD 128

typedef __bf16 bf16;
typedef __attribute__((ext_vector_type(16))) __bf16       v16bf;
typedef __attribute__((ext_vector_type(8)))  __bf16       bf16x8;
typedef __attribute__((ext_vector_type(4)))  __bf16       bf16x4;
typedef __attribute__((ext_vector_type(8)))  float        v8f;
typedef __attribute__((ext_vector_type(4)))  float        v4f;
typedef __attribute__((ext_vector_type(4)))  unsigned int v4u;
typedef __attribute__((ext_vector_type(4)))  int          v4i;

static __device__ __forceinline__ v16bf make_frag(const bf16* p_lo, const bf16* p_hi) {
  bf16x8 lo = *reinterpret_cast<const bf16x8*>(p_lo);
  bf16x8 hi = *reinterpret_cast<const bf16x8*>(p_hi);
  return __builtin_shufflevector(lo, hi, 0,1,2,3,4,5,6,7,8,9,10,11,12,13,14,15);
}

static __device__ __forceinline__ v8f v8f_zero() {
  v8f z;
  #pragma unroll
  for (int i = 0; i < 8; ++i) z[i] = 0.0f;
  return z;
}

// ---- CDNA5 async global->LDS copy (ASYNCcnt path), with safe fallbacks ----
static __device__ __forceinline__ void async_copy_b128(const bf16* gsrc, bf16* ldst) {
#if __has_builtin(__builtin_amdgcn_global_load_async_to_lds_b128)
  __builtin_amdgcn_global_load_async_to_lds_b128(
      (__attribute__((address_space(1))) v4i*)(gsrc),
      (__attribute__((address_space(3))) v4i*)(ldst), 0, 0);
#else
  *reinterpret_cast<v4u*>(ldst) = *reinterpret_cast<const v4u*>(gsrc);
#endif
}

static __device__ __forceinline__ void wait_async0() {
#if __has_builtin(__builtin_amdgcn_s_wait_asynccnt)
  __builtin_amdgcn_s_wait_asynccnt(0);
#else
  asm volatile("s_wait_asynccnt 0x0" ::: "memory");
#endif
}

// ---------------------------------------------------------------------------
// Kernel 1: Q/K/V = x @ W^T   (f32 in, bf16 out to workspace)
// Block = 128 threads (4 waves); each block does 64 rows of flattened (B*S).
// ---------------------------------------------------------------------------
static __device__ __forceinline__ void project_tile(
    const float* __restrict__ W, bf16* __restrict__ Out, bf16* ldsW,
    const v16bf aX[4], int tid, int lane, int mrow, int khalf,
    int rowTile, int wave)
{
  __syncthreads();
  {
    const v4f* wsrc = reinterpret_cast<const v4f*>(W);
    for (int i = tid; i < (DD * DD) / 4; i += 128) {
      v4f f = wsrc[i];
      bf16x4 h;
      h.x = (bf16)f.x; h.y = (bf16)f.y; h.z = (bf16)f.z; h.w = (bf16)f.w;
      *reinterpret_cast<bf16x4*>(ldsW + i * 4) = h;
    }
  }
  __syncthreads();

  bf16* outp = Out + (size_t)(rowTile + wave * 16) * DD;
  for (int e0 = 0; e0 < 8; ++e0) {            // output-column subtiles of 16
    v8f c = v8f_zero();
    #pragma unroll
    for (int kb = 0; kb < 4; ++kb) {
      // B[k][n] = W[e0*16 + n][k] : lane's column = W row, contiguous reads
      const bf16* base = ldsW + (e0 * 16 + mrow) * DD + kb * 32;
      v16bf bw = make_frag(base + khalf, base + 16 + khalf);
      c = __builtin_amdgcn_wmma_f32_16x16x32_bf16(
              false, aX[kb], false, bw, (short)0, c, false, false);
    }
    const int col = e0 * 16 + mrow;
    #pragma unroll
    for (int r = 0; r < 8; ++r) {
      const int mr = (lane < 16) ? r : (r + 8);
      outp[(size_t)mr * DD + col] = (bf16)c[r];
    }
  }
}

__global__ __launch_bounds__(128) void qkv_kernel(
    const float* __restrict__ x,  const float* __restrict__ Wq,
    const float* __restrict__ Wk, const float* __restrict__ Wv,
    bf16* __restrict__ Q, bf16* __restrict__ K, bf16* __restrict__ V)
{
  __shared__ __align__(16) bf16 ldsX[64 * DD];   // 16 KB, x tile bf16
  __shared__ __align__(16) bf16 ldsW[DD * DD];   // 32 KB, W bf16

  const int tid     = threadIdx.x;
  const int wave    = tid >> 5;
  const int lane    = tid & 31;
  const int mrow    = lane & 15;
  const int khalf   = (lane < 16) ? 0 : 8;
  const int rowTile = blockIdx.x * 64;           // base row in flattened (B*S)

  // stage x tile as bf16 (native cvt)
  {
    const v4f* src = reinterpret_cast<const v4f*>(x + (size_t)rowTile * DD);
    for (int i = tid; i < (64 * DD) / 4; i += 128) {
      v4f f = src[i];
      bf16x4 h;
      h.x = (bf16)f.x; h.y = (bf16)f.y; h.z = (bf16)f.z; h.w = (bf16)f.w;
      *reinterpret_cast<bf16x4*>(ldsX + i * 4) = h;
    }
  }
  __syncthreads();

  // this wave's A fragments: 16 rows x 128 cols, 4 fragments of 16x32
  v16bf aX[4];
  #pragma unroll
  for (int kb = 0; kb < 4; ++kb) {
    const bf16* base = ldsX + (wave * 16 + mrow) * DD + kb * 32;
    aX[kb] = make_frag(base + khalf, base + 16 + khalf);
  }

  // three separate inlined calls keep pointers concrete -> global stores
  project_tile(Wq, Q, ldsW, aX, tid, lane, mrow, khalf, rowTile, wave);
  project_tile(Wk, K, ldsW, aX, tid, lane, mrow, khalf, rowTile, wave);
  project_tile(Wv, V, ldsW, aX, tid, lane, mrow, khalf, rowTile, wave);
}

// ---------------------------------------------------------------------------
// Kernel 2: causal flash attention over bf16 Q/K/V, f32 accumulate/output.
// Grid (S/64, B); block = 128 threads (4 waves); wave owns 16 query rows.
// ---------------------------------------------------------------------------
__global__ __launch_bounds__(128) void attn_kernel(
    const bf16* __restrict__ Q, const bf16* __restrict__ K,
    const bf16* __restrict__ V, float* __restrict__ out)
{
  __shared__ __align__(16) bf16 ldsK [64 * DD];     // 16 KB [krow][d]
  __shared__ __align__(16) bf16 ldsVt[DD * 64];     // 16 KB [d][krow]
  __shared__ __align__(16) bf16 ldsP [4 * 16 * 64]; //  8 KB per-wave P

  const int tid   = threadIdx.x;
  const int wave  = tid >> 5;
  const int lane  = tid & 31;
  const int mrow  = lane & 15;
  const int khalf = (lane < 16) ? 0 : 8;
  const int b     = blockIdx.y;
  const int qtile = blockIdx.x * 64;
  const float scale = 0.08838834764831845f;   // 1/sqrt(128)
  const float LOG2E = 1.4426950408889634f;

  // Q fragments for this wave's 16 rows (A-layout, straight from global bf16)
  v16bf aQ[4];
  {
    const bf16* qbase = Q + (size_t)(b * SS + qtile + wave * 16 + mrow) * DD;
    #pragma unroll
    for (int kb = 0; kb < 4; ++kb)
      aQ[kb] = make_frag(qbase + kb * 32 + khalf, qbase + kb * 32 + 16 + khalf);
  }

  v8f o[8];
  #pragma unroll
  for (int i = 0; i < 8; ++i) o[i] = v8f_zero();
  float mstat[8], lstat[8];
  #pragma unroll
  for (int r = 0; r < 8; ++r) { mstat[r] = -3.0e38f; lstat[r] = 0.0f; }

  const int jmax = blockIdx.x;   // causal: key tiles 0..jmax
  for (int j = 0; j <= jmax; ++j) {
    const int jbase = j * 64;
    __syncthreads();   // previous iteration's LDS reads complete

    // stage K tile row-major via async global->LDS (ASYNCcnt path)
    {
      const bf16* ksrc = K + (size_t)(b * SS + jbase) * DD;
      for (int i = tid; i < (64 * DD) / 8; i += 128)
        async_copy_b128(ksrc + i * 8, ldsK + i * 8);
    }
    // stage V tile transposed (needs VGPR round trip): ldsVt[d][krow]
    {
      const bf16* vsrc = V + (size_t)(b * SS + jbase) * DD;
      for (int idx = tid; idx < 64 * 16; idx += 128) {
        const int krow = idx >> 4;
        const int cg   = idx & 15;
        union { v4u v; bf16 s[8]; } u;
        u.v = *reinterpret_cast<const v4u*>(vsrc + (size_t)krow * DD + cg * 8);
        #pragma unroll
        for (int e = 0; e < 8; ++e)
          ldsVt[(cg * 8 + e) * 64 + krow] = u.s[e];
      }
    }
    wait_async0();
    __syncthreads();

    // prefetch next key tile into L2/L0 while we compute on this one
    if (j < jmax) {
      const bf16* kn = K + (size_t)(b * SS + jbase + 64) * DD;
      const bf16* vn = V + (size_t)(b * SS + jbase + 64) * DD;
      __builtin_prefetch(kn + tid * 64, 0, 0);   // 128B per lane
      __builtin_prefetch(vn + tid * 64, 0, 0);
    }

    // S = Q K^T : 16x64 per wave (4 subtiles of 16x16)
    v8f c[4];
    for (int n = 0; n < 4; ++n) {
      c[n] = v8f_zero();
      #pragma unroll
      for (int kb = 0; kb < 4; ++kb) {
        // B[k][n] = K[n0+n][k]: lane's column = K row (contiguous in ldsK)
        const bf16* base = ldsK + (n * 16 + mrow) * DD + kb * 32;
        v16bf bk = make_frag(base + khalf, base + 16 + khalf);
        c[n] = __builtin_amdgcn_wmma_f32_16x16x32_bf16(
                   false, aQ[kb], false, bk, (short)0, c[n], false, false);
      }
    }

    // scale + causal mask + per-row tile max
    const bool diag = (j == jmax);
    float tmax[8];
    #pragma unroll
    for (int r = 0; r < 8; ++r) tmax[r] = -3.0e38f;
    for (int n = 0; n < 4; ++n) {
      const int kcol = jbase + n * 16 + mrow;
      #pragma unroll
      for (int r = 0; r < 8; ++r) {
        float v = c[n][r] * scale;
        if (diag) {
          const int qrow = qtile + wave * 16 + ((lane < 16) ? r : r + 8);
          if (kcol > qrow) v = -3.0e38f;
        }
        c[n][r] = v;
        tmax[r] = fmaxf(tmax[r], v);
      }
    }
    // reduce max across the 16-lane half (xor masks < 16 never cross halves)
    #pragma unroll
    for (int r = 0; r < 8; ++r) {
      float t = tmax[r];
      t = fmaxf(t, __shfl_xor(t, 1));
      t = fmaxf(t, __shfl_xor(t, 2));
      t = fmaxf(t, __shfl_xor(t, 4));
      t = fmaxf(t, __shfl_xor(t, 8));
      tmax[r] = t;
    }

    float alpha[8], psum[8];
    #pragma unroll
    for (int r = 0; r < 8; ++r) {
      const float mnew = fmaxf(mstat[r], tmax[r]);
      alpha[r] = exp2f((mstat[r] - mnew) * LOG2E);
      mstat[r] = mnew;
      psum[r]  = 0.0f;
    }
    for (int n = 0; n < 4; ++n) {
      #pragma unroll
      for (int r = 0; r < 8; ++r) {
        const float p = exp2f((c[n][r] - mstat[r]) * LOG2E);
        c[n][r] = p;
        psum[r] += p;
      }
    }
    #pragma unroll
    for (int r = 0; r < 8; ++r) {
      float t = psum[r];
      t += __shfl_xor(t, 1);
      t += __shfl_xor(t, 2);
      t += __shfl_xor(t, 4);
      t += __shfl_xor(t, 8);
      lstat[r] = lstat[r] * alpha[r] + t;
    }
    // rescale running O
    for (int i = 0; i < 8; ++i) {
      #pragma unroll
      for (int r = 0; r < 8; ++r) o[i][r] *= alpha[r];
    }

    // re-layout P: C-layout regs -> row-major bf16 in per-wave LDS region
    bf16* pw = ldsP + wave * (16 * 64);
    for (int n = 0; n < 4; ++n) {
      #pragma unroll
      for (int r = 0; r < 8; ++r) {
        const int prow = (lane < 16) ? r : (r + 8);
        pw[prow * 64 + n * 16 + mrow] = (bf16)c[n][r];
      }
    }
    // same wave reads its own region back: no cross-wave barrier needed
    v16bf aP[2];
    #pragma unroll
    for (int kb2 = 0; kb2 < 2; ++kb2) {
      const bf16* base = pw + mrow * 64 + kb2 * 32;
      aP[kb2] = make_frag(base + khalf, base + 16 + khalf);
    }

    // O += P x V  (B[k][n] = V[k][n0+n] = ldsVt[n0+n][k], contiguous)
    for (int n = 0; n < 8; ++n) {
      #pragma unroll
      for (int kb2 = 0; kb2 < 2; ++kb2) {
        const bf16* base = ldsVt + (n * 16 + mrow) * 64 + kb2 * 32;
        v16bf bv = make_frag(base + khalf, base + 16 + khalf);
        o[n] = __builtin_amdgcn_wmma_f32_16x16x32_bf16(
                   false, aP[kb2], false, bv, (short)0, o[n], false, false);
      }
    }
  }

  // normalize and store f32 output
  for (int n = 0; n < 8; ++n) {
    const int col = n * 16 + mrow;
    #pragma unroll
    for (int r = 0; r < 8; ++r) {
      const int qrow = qtile + wave * 16 + ((lane < 16) ? r : r + 8);
      out[(size_t)(b * SS + qrow) * DD + col] = o[n][r] * (1.0f / lstat[r]);
    }
  }
}

// ---------------------------------------------------------------------------
extern "C" void kernel_launch(void* const* d_in, const int* in_sizes, int n_in,
                              void* d_out, int out_size, void* d_ws, size_t ws_size,
                              hipStream_t stream) {
  const float* x  = (const float*)d_in[0];
  const float* Wq = (const float*)d_in[1];
  const float* Wk = (const float*)d_in[2];
  const float* Wv = (const float*)d_in[3];
  float* out = (float*)d_out;

  // workspace: bf16 Q, K, V (B*S*D each = 4 MiB each, 12 MiB total)
  bf16* Qw = (bf16*)d_ws;
  bf16* Kw = Qw + (size_t)BB * SS * DD;
  bf16* Vw = Kw + (size_t)BB * SS * DD;

  qkv_kernel<<<dim3((BB * SS) / 64), dim3(128), 0, stream>>>(x, Wq, Wk, Wv, Qw, Kw, Vw);
  attn_kernel<<<dim3(SS / 64, BB), dim3(128), 0, stream>>>(Qw, Kw, Vw, out);

  (void)in_sizes; (void)n_in; (void)out_size; (void)ws_size;
}
